// MultiDCRNN_4449586119220
// MI455X (gfx1250) — compile-verified, matrix operand order checked by reading, not run on hardware
//
#include <hip/hip_runtime.h>
#include <hip/hip_bf16.h>
#include <math.h>

// ---------------------------------------------------------------------------
// MultiDCRNN GRU cell for gfx1250 (MI455X).
//  - sparse diffusion: f32 global atomics (bandwidth-bound part)
//  - dense projections: single K=480 GEMM per dconv via v_wmma_f32_16x16x32_f16
//  - B matrix staged block-wide in LDS via global_load_async_to_lds_b128
//  - gate epilogues fused into the GEMM epilogue
// ---------------------------------------------------------------------------

typedef __attribute__((ext_vector_type(16))) _Float16 v16h;
typedef __attribute__((ext_vector_type(8)))  float    v8f;

#define B_      2
#define N_S_    20000
#define N_P_    8000
#define NT_     28000      // N_S_ + N_P_
#define E_S_    320000
#define E_P_    128000
#define INC_    32
#define OUTC_   64
#define CCH_    96         // INC_ + OUTC_
#define KTOT_   480        // 5 * CCH_
#define KTILES_ 15         // KTOT_ / 32
#define BFRAG_HALVES_ (KTILES_ * 4 * 32 * 16)   // 30720 halves = 61440 bytes

static __device__ __forceinline__ float sigmoidf_(float x) {
  return 1.0f / (1.0f + expf(-x));
}

// ---------------------------------------------------------------- utilities
__global__ void k_zero(float* __restrict__ p, size_t n) {
  size_t i = (size_t)blockIdx.x * blockDim.x + threadIdx.x;
  if (i < n) p[i] = 0.0f;
}

// ------------------------------------------------------- degree / coefficient
__global__ void k_deg(const int* __restrict__ ei, const float* __restrict__ w,
                      int E, float* __restrict__ deg_out, float* __restrict__ deg_in) {
  int e = blockIdx.x * blockDim.x + threadIdx.x;
  if (e >= E) return;
  atomicAdd(&deg_out[ei[e]],     w[e]);  // row
  atomicAdd(&deg_in [ei[E + e]], w[e]);  // col
}

__global__ void k_coef(const int* __restrict__ ei, const float* __restrict__ w,
                       int E, const float* __restrict__ deg_out,
                       const float* __restrict__ deg_in,
                       float* __restrict__ coef_f, float* __restrict__ coef_b) {
  int e = blockIdx.x * blockDim.x + threadIdx.x;
  if (e >= E) return;
  coef_f[e] = w[e] / deg_out[ei[e]];
  coef_b[e] = w[e] / deg_in [ei[E + e]];
}

// --------------------------------------------------------------- concat X|H
__global__ void k_concat(const float* __restrict__ X, const float* __restrict__ Hfull,
                         int N, int nodeOff, float* __restrict__ C) {
  size_t idx = (size_t)blockIdx.x * blockDim.x + threadIdx.x;
  size_t total = (size_t)B_ * N * CCH_;
  if (idx >= total) return;
  int c = (int)(idx % CCH_);
  size_t bn = idx / CCH_;
  int n = (int)(bn % N);
  int b = (int)(bn / N);
  float v = (c < INC_)
      ? X[((size_t)b * N + n) * INC_ + c]
      : Hfull[((size_t)b * NT_ + nodeOff + n) * OUTC_ + (c - INC_)];
  C[idx] = v;
}

// ---------------------------------------------------- sparse diffusion step
// dst[:, dst_idx[e], :] += coef[e] * src[:, src_idx[e], :]
__global__ void k_diffuse(const float* __restrict__ src_feat, float* __restrict__ dst_feat,
                          const int* __restrict__ src_idx, const int* __restrict__ dst_idx,
                          const float* __restrict__ coef, int E, int N) {
  int idx = blockIdx.x * blockDim.x + threadIdx.x;
  int total = E * B_ * 6;                 // 6 groups of 16 channels
  if (idx >= total) return;
  int cg = idx % 6;
  int eb = idx / 6;
  int e  = eb % E;
  int b  = eb / E;
  float cw = coef[e];
  const float* sp = src_feat + ((size_t)b * N + src_idx[e]) * CCH_ + cg * 16;
  float*       dp = dst_feat + ((size_t)b * N + dst_idx[e]) * CCH_ + cg * 16;
#pragma unroll
  for (int c = 0; c < 16; ++c) atomicAdd(&dp[c], cw * sp[c]);
}

// -------------------------------------- pack [C|T1o|T1i|2T2o-C|2T2i-C] -> f16
__global__ void k_pack(const float* __restrict__ C,  const float* __restrict__ T1o,
                       const float* __restrict__ T1i, const float* __restrict__ T2o,
                       const float* __restrict__ T2i, _Float16* __restrict__ A, int M) {
  size_t idx = (size_t)blockIdx.x * blockDim.x + threadIdx.x;
  size_t total = (size_t)M * (KTOT_ / 8);
  if (idx >= total) return;
  int g = (int)(idx % (KTOT_ / 8));
  size_t m = idx / (KTOT_ / 8);
  int kk = g * 8;                 // 8 consecutive K, all within one tensor block
  int t  = kk / CCH_;
  int ch = kk % CCH_;
  size_t o = m * CCH_ + ch;
  _Float16* ap = A + m * KTOT_ + kk;
#pragma unroll
  for (int j = 0; j < 8; ++j) {
    float v;
    if      (t == 0) v = C[o + j];
    else if (t == 1) v = T1o[o + j];
    else if (t == 2) v = T1i[o + j];
    else if (t == 3) v = 2.0f * T2o[o + j] - C[o + j];
    else             v = 2.0f * T2i[o + j] - C[o + j];
    ap[j] = (_Float16)v;
  }
}

// --------------------------- pre-swizzle W[2,3,96,64] into B fragment layout
// Bf[((kt*4+nt)*32+lane)*16+h]; 16-bit B 32x16 lane layout mirrors A layout.
__global__ void k_prepB(const float* __restrict__ W, _Float16* __restrict__ Bf) {
  int idx = blockIdx.x * blockDim.x + threadIdx.x;
  if (idx >= BFRAG_HALVES_) return;
  int h    = idx & 15;
  int lane = (idx >> 4) & 31;
  int nt   = (idx >> 9) & 3;
  int kt   = idx >> 11;
  int col  = nt * 16 + (lane & 15);
  int kb   = (lane < 16) ? 0 : 8;
  int koff = (h < 8) ? (kb + h) : (16 + kb + (h - 8));
  int k    = kt * 32 + koff;
  int t    = k / CCH_;
  int ch   = k % CCH_;
  float w;
#define WAT(a, c) W[((((a) * 3) + (c)) * CCH_ + ch) * OUTC_ + col]
  if      (t == 0) w = WAT(0, 0) + WAT(1, 0);
  else if (t == 1) w = WAT(0, 1);
  else if (t == 2) w = WAT(1, 1);
  else if (t == 3) w = WAT(0, 2);
  else             w = WAT(1, 2);
#undef WAT
  Bf[idx] = (_Float16)w;
}

// -------------------------------------------------------------- WMMA GEMM
// One wave -> one 16-row x 64-col output tile; K loop of 15 x 32.
// B fragments staged once per block in LDS (async DMA), served by ds_load.
// mode 0: gateOut = sigmoid(acc + bias)
// mode 1: finalOut = Z*H + (1-Z)*tanh(acc + bias)
union AFragU { v16h v; float4 f[2]; };

__global__ __launch_bounds__(256) void k_gemm(
    const _Float16* __restrict__ A, const _Float16* __restrict__ Bfrag,
    const float* __restrict__ bias, int M, int N, int nodeOff, int mode,
    float* __restrict__ gateOut, const float* __restrict__ Zbuf,
    const float* __restrict__ Hin, float* __restrict__ finalOut) {
  __shared__ __attribute__((aligned(32))) _Float16 smemB[BFRAG_HALVES_];

  // ---- cooperative async DMA: global Bfrag -> LDS (16 B per lane per op) ----
  {
    unsigned int ldsBase = (unsigned int)(uintptr_t)(&smemB[0]);
#pragma unroll
    for (int i = 0; i < BFRAG_HALVES_ * 2 / (256 * 16); ++i) {   // 15 chunks/thread
      unsigned int byteOff = (threadIdx.x + i * 256) * 16;
      asm volatile("global_load_async_to_lds_b128 %0, %1, off"
                   :: "v"(ldsBase + byteOff),
                      "v"((const char*)Bfrag + byteOff)
                   : "memory");
    }
    asm volatile("s_wait_asynccnt 0x0" ::: "memory");
  }
  __syncthreads();

  int wave = threadIdx.x >> 5;
  int lane = threadIdx.x & 31;
  int tile = blockIdx.x * (blockDim.x >> 5) + wave;
  int mBase = tile * 16;
  if (mBase >= M) return;   // tail waves: after barrier, safe to exit

  v8f acc0 = {}, acc1 = {}, acc2 = {}, acc3 = {};
  const int kbase = (lane < 16) ? 0 : 8;                      // 16-bit A layout
  const _Float16* arow = A + (size_t)(mBase + (lane & 15)) * KTOT_;
  const int bLane = lane * 16;

  for (int kt = 0; kt < KTILES_; ++kt) {
    AFragU a;
    a.f[0] = *(const float4*)(arow + kt * 32 + kbase);        // K  0.. 7 (+kb)
    a.f[1] = *(const float4*)(arow + kt * 32 + kbase + 16);   // K 16..23 (+kb)
    int bkt = kt * 2048 + bLane;                              // kt*4*32*16
    v16h b0 = *(const v16h*)(&smemB[bkt + 0 * 512]);
    v16h b1 = *(const v16h*)(&smemB[bkt + 1 * 512]);
    v16h b2 = *(const v16h*)(&smemB[bkt + 2 * 512]);
    v16h b3 = *(const v16h*)(&smemB[bkt + 3 * 512]);
    acc0 = __builtin_amdgcn_wmma_f32_16x16x32_f16(false, a.v, false, b0, (short)0, acc0, false, false);
    acc1 = __builtin_amdgcn_wmma_f32_16x16x32_f16(false, a.v, false, b1, (short)0, acc1, false, false);
    acc2 = __builtin_amdgcn_wmma_f32_16x16x32_f16(false, a.v, false, b2, (short)0, acc2, false, false);
    acc3 = __builtin_amdgcn_wmma_f32_16x16x32_f16(false, a.v, false, b3, (short)0, acc3, false, false);
  }

  // D layout: VGPR i -> row i (lanes 0-15) / row i+8 (lanes 16-31); col = lane%16
  int mloc = (lane < 16) ? 0 : 8;
  int colb = lane & 15;
#pragma unroll
  for (int i = 0; i < 8; ++i) {
    int m = mBase + mloc + i;
    int b = m / N, n = m % N;
    size_t ob = ((size_t)b * NT_ + nodeOff + n) * OUTC_;
    float vv[4] = {acc0[i], acc1[i], acc2[i], acc3[i]};
#pragma unroll
    for (int nt = 0; nt < 4; ++nt) {
      int col = colb + nt * 16;
      float v = vv[nt] + bias[col];
      if (mode == 0) {
        gateOut[ob + col] = sigmoidf_(v);
      } else {
        float ht = tanhf(v);
        float z  = Zbuf[ob + col];
        finalOut[ob + col] = z * Hin[ob + col] + (1.0f - z) * ht;
      }
    }
  }
}

// ------------------------------------------------------------ gated hidden
__global__ void k_gh(const float* __restrict__ H, const float* __restrict__ R,
                     float* __restrict__ gH, size_t n) {
  size_t i = (size_t)blockIdx.x * blockDim.x + threadIdx.x;
  if (i < n) gH[i] = H[i] * R[i];
}

// ---------------------------------------------------------------------------
static inline unsigned int nblk(size_t n, int b) {
  return (unsigned int)((n + (size_t)b - 1) / (size_t)b);
}

extern "C" void kernel_launch(void* const* d_in, const int* in_sizes, int n_in,
                              void* d_out, int out_size, void* d_ws, size_t ws_size,
                              hipStream_t stream) {
  (void)in_sizes; (void)n_in; (void)out_size; (void)ws_size;

  const float* X_s  = (const float*)d_in[0];
  const float* X_p  = (const float*)d_in[1];
  const int*   ei_s = (const int*)  d_in[2];
  const float* ew_s = (const float*)d_in[3];
  const int*   ei_p = (const int*)  d_in[4];
  const float* ew_p = (const float*)d_in[5];
  // d_in[6], d_in[7]: p2s edges — unused by the reference
  const float* H    = (const float*)d_in[8];
  const float* Wz_s = (const float*)d_in[9];  const float* bz_s = (const float*)d_in[10];
  const float* Wz_p = (const float*)d_in[11]; const float* bz_p = (const float*)d_in[12];
  const float* Wr_s = (const float*)d_in[13]; const float* br_s = (const float*)d_in[14];
  const float* Wr_p = (const float*)d_in[15]; const float* br_p = (const float*)d_in[16];
  const float* Wh_s = (const float*)d_in[17]; const float* bh_s = (const float*)d_in[18];
  const float* Wh_p = (const float*)d_in[19]; const float* bh_p = (const float*)d_in[20];
  float* out = (float*)d_out;

  // ---- workspace carve-up (~162 MB) ----
  char* wsb = (char*)d_ws; size_t off = 0;
  auto alloc = [&](size_t bytes) -> void* {
    void* p = wsb + off; off = (off + bytes + 255) & ~(size_t)255; return p;
  };
  float* deg_os = (float*)alloc((size_t)N_S_ * 4);
  float* deg_is = (float*)alloc((size_t)N_S_ * 4);
  float* deg_op = (float*)alloc((size_t)N_P_ * 4);
  float* deg_ip = (float*)alloc((size_t)N_P_ * 4);
  float* cf_s   = (float*)alloc((size_t)E_S_ * 4);
  float* cb_s   = (float*)alloc((size_t)E_S_ * 4);
  float* cf_p   = (float*)alloc((size_t)E_P_ * 4);
  float* cb_p   = (float*)alloc((size_t)E_P_ * 4);
  size_t featS  = (size_t)B_ * N_S_ * CCH_;            // street-sized, reused for POI
  float* Cbuf = (float*)alloc(featS * 4);
  float* T1o  = (float*)alloc(featS * 4);
  float* T1i  = (float*)alloc(featS * 4);
  float* T2o  = (float*)alloc(featS * 4);
  float* T2i  = (float*)alloc(featS * 4);
  _Float16* Apack = (_Float16*)alloc((size_t)B_ * N_S_ * KTOT_ * 2);
  _Float16* Bfrag = (_Float16*)alloc((size_t)6 * BFRAG_HALVES_ * 2);
  size_t nH = (size_t)B_ * NT_ * OUTC_;
  float* Zbuf = (float*)alloc(nH * 4);
  float* Rbuf = (float*)alloc(nH * 4);
  float* gH   = (float*)alloc(nH * 4);

  // ---- degrees & edge coefficients ----
  k_zero<<<nblk(N_S_, 256), 256, 0, stream>>>(deg_os, N_S_);
  k_zero<<<nblk(N_S_, 256), 256, 0, stream>>>(deg_is, N_S_);
  k_zero<<<nblk(N_P_, 256), 256, 0, stream>>>(deg_op, N_P_);
  k_zero<<<nblk(N_P_, 256), 256, 0, stream>>>(deg_ip, N_P_);
  k_deg <<<nblk(E_S_, 256), 256, 0, stream>>>(ei_s, ew_s, E_S_, deg_os, deg_is);
  k_deg <<<nblk(E_P_, 256), 256, 0, stream>>>(ei_p, ew_p, E_P_, deg_op, deg_ip);
  k_coef<<<nblk(E_S_, 256), 256, 0, stream>>>(ei_s, ew_s, E_S_, deg_os, deg_is, cf_s, cb_s);
  k_coef<<<nblk(E_P_, 256), 256, 0, stream>>>(ei_p, ew_p, E_P_, deg_op, deg_ip, cf_p, cb_p);

  // ---- pre-swizzled f16 B fragments (order: z_s z_p r_s r_p h_s h_p) ----
  const float* Ws[6] = {Wz_s, Wz_p, Wr_s, Wr_p, Wh_s, Wh_p};
  for (int i = 0; i < 6; ++i)
    k_prepB<<<nblk(BFRAG_HALVES_, 256), 256, 0, stream>>>(Ws[i], Bfrag + (size_t)i * BFRAG_HALVES_);

  const int    Ns[2]   = {N_S_, N_P_};
  const int    Es[2]   = {E_S_, E_P_};
  const int    offs[2] = {0, N_S_};
  const float* Xs[2]   = {X_s, X_p};
  const int*   eis[2]  = {ei_s, ei_p};
  float*       cfs[2]  = {cf_s, cf_p};
  float*       cbs[2]  = {cb_s, cb_p};
  const float* bz[2]   = {bz_s, bz_p};
  const float* br[2]   = {br_s, br_p};
  const float* bh[2]   = {bh_s, bh_p};

  for (int phase = 0; phase < 2; ++phase) {
    const float* Hsrc = (phase == 0) ? H : gH;
    for (int g = 0; g < 2; ++g) {
      int N = Ns[g], E = Es[g], noff = offs[g];
      size_t feat = (size_t)B_ * N * CCH_;
      const int* row = eis[g];      // edge_index[0]
      const int* col = eis[g] + E;  // edge_index[1]

      k_concat<<<nblk(feat, 256), 256, 0, stream>>>(Xs[g], Hsrc, N, noff, Cbuf);
      k_zero<<<nblk(feat, 256), 256, 0, stream>>>(T1o, feat);
      k_zero<<<nblk(feat, 256), 256, 0, stream>>>(T1i, feat);
      k_zero<<<nblk(feat, 256), 256, 0, stream>>>(T2o, feat);
      k_zero<<<nblk(feat, 256), 256, 0, stream>>>(T2i, feat);

      int dif = E * B_ * 6;
      // forward: src=col, dst=row, coef_fwd ; reverse: src=row, dst=col, coef_bwd
      k_diffuse<<<nblk(dif, 256), 256, 0, stream>>>(Cbuf, T1o, col, row, cfs[g], E, N);
      k_diffuse<<<nblk(dif, 256), 256, 0, stream>>>(Cbuf, T1i, row, col, cbs[g], E, N);
      k_diffuse<<<nblk(dif, 256), 256, 0, stream>>>(T1o,  T2o, col, row, cfs[g], E, N);
      k_diffuse<<<nblk(dif, 256), 256, 0, stream>>>(T1i,  T2i, row, col, cbs[g], E, N);

      int M = B_ * N;
      k_pack<<<nblk((size_t)M * (KTOT_ / 8), 256), 256, 0, stream>>>(
          Cbuf, T1o, T1i, T2o, T2i, Apack, M);

      int tiles = M / 16;                    // M is a multiple of 16
      unsigned int gb = (tiles + 7) / 8;     // 8 waves / block
      if (phase == 0) {
        k_gemm<<<gb, 256, 0, stream>>>(Apack, Bfrag + (size_t)(0 * 2 + g) * BFRAG_HALVES_,
                                       bz[g], M, N, noff, 0, Zbuf, nullptr, nullptr, nullptr);
        k_gemm<<<gb, 256, 0, stream>>>(Apack, Bfrag + (size_t)(1 * 2 + g) * BFRAG_HALVES_,
                                       br[g], M, N, noff, 0, Rbuf, nullptr, nullptr, nullptr);
      } else {
        k_gemm<<<gb, 256, 0, stream>>>(Apack, Bfrag + (size_t)(2 * 2 + g) * BFRAG_HALVES_,
                                       bh[g], M, N, noff, 1, nullptr, Zbuf, H, out);
      }
    }
    if (phase == 0)
      k_gh<<<nblk(nH, 256), 256, 0, stream>>>(H, Rbuf, gH, nH);
  }
}
